// MojoFusedLinearCrossEntropyLoss_1065151889688
// MI455X (gfx1250) — matrix operand order repaired and checked by reading, not compiled, not run on hardware
//
#include <hip/hip_runtime.h>
#include <math.h>
#include <stdint.h>

typedef __attribute__((ext_vector_type(16))) _Float16 v16h;
typedef __attribute__((ext_vector_type(8)))  _Float16 v8h;
typedef __attribute__((ext_vector_type(4)))  _Float16 v4h;
typedef __attribute__((ext_vector_type(8)))  float    v8f;

#define N_TOK 8192
#define HID   2048
#define VOC   32000

#define BT 128               // tokens per workgroup
#define BV 128               // vocab rows per workgroup
#define BK 32                // K step (f16 WMMA K)
#define LDK 40               // padded LDS row stride (halves) -> 80B rows
#define VBLKS (VOC / BV)     // 250
#define TBLKS (N_TOK / BT)   // 64
#define NKIT  (HID / BK)     // 64

#define L2E 1.4426950408889634f
#define LN2 0.6931471805599453f

// ---------------------------------------------------------------------------
// CDNA5 async copy helpers (ASYNCcnt-tracked global->LDS, no VGPR data path)
// ---------------------------------------------------------------------------
__device__ __forceinline__ void async_g2l_b128(uint32_t lds_off, const void* gaddr) {
  // GLOBAL_LOAD_ASYNC_TO_LDS_B128: vdst = LDS byte address, vaddr = 64-bit VA
  asm volatile("global_load_async_to_lds_b128 %0, %1, off"
               :: "v"(lds_off), "v"(gaddr) : "memory");
}

__device__ __forceinline__ void wait_async0() {
#if __has_builtin(__builtin_amdgcn_s_wait_asynccnt)
  __builtin_amdgcn_s_wait_asynccnt(0);
#else
  asm volatile("s_wait_asynccnt 0x0" ::: "memory");
#endif
}

__device__ __forceinline__ uint32_t lds_byte_off(const void* p) {
  // flat shared-aperture pointers carry the wave-relative LDS offset in [31:0]
  return (uint32_t)(uintptr_t)p;
}

// ---------------------------------------------------------------------------
// Shared tile-compute: 8 WMMA per K-step per wave (2 vocab x 4 token tiles)
// A (weight 16x32 f16): lanes<16 K{0..7,16..23}, lanes>=16 K{8..15,24..31}
// B (input 32x16 f16):  lanes<16 K{0..7,16..23}, lanes>=16 K{8..15,24..31}
// ---------------------------------------------------------------------------
__device__ __forceinline__ void do_tile(const _Float16* sWb, const _Float16* sXb,
                                        int wm, int wn, int l16, int half,
                                        v8f acc[2][4])
{
  v16h afr[2];
  #pragma unroll
  for (int mt = 0; mt < 2; ++mt) {
    const _Float16* p = &sWb[(wm * 32 + mt * 16 + l16) * LDK + half * 8];
    v8h lo = *(const v8h*)p;
    v8h hi = *(const v8h*)(p + 16);
    #pragma unroll
    for (int i = 0; i < 8; ++i) { afr[mt][i] = lo[i]; afr[mt][8 + i] = hi[i]; }
  }
  v16h bfr[4];
  #pragma unroll
  for (int nt = 0; nt < 4; ++nt) {
    const _Float16* p = &sXb[(wn * 64 + nt * 16 + l16) * LDK + half * 8];
    v8h lo = *(const v8h*)p;
    v8h hi = *(const v8h*)(p + 16);
    #pragma unroll
    for (int i = 0; i < 8; ++i) { bfr[nt][i] = lo[i]; bfr[nt][8 + i] = hi[i]; }
  }
  #pragma unroll
  for (int mt = 0; mt < 2; ++mt)
    #pragma unroll
    for (int nt = 0; nt < 4; ++nt)
      acc[mt][nt] = __builtin_amdgcn_wmma_f32_16x16x32_f16(
          false, afr[mt], false, bfr[nt], (short)0, acc[mt][nt], false, false);
}

// ---------------------------------------------------------------------------
// Shared epilogue: bias + per-lane partial softmax + cross-wave LDS combine
// ---------------------------------------------------------------------------
__device__ __forceinline__ void epilogue(v8f acc[2][4], int vblk, int tblk,
                                         int wm, int wn, int l16, int half, int tid,
                                         const long long* __restrict__ TGT,
                                         const float* __restrict__ BIAS,
                                         float* __restrict__ pm, float* __restrict__ ps,
                                         float* __restrict__ tl,
                                         float* red_m, float* red_s)
{
  #pragma unroll
  for (int mt = 0; mt < 2; ++mt) {
    const int mrow = vblk * BV + wm * 32 + mt * 16 + half * 8;
    float4 b0 = *(const float4*)(BIAS + mrow);
    float4 b1 = *(const float4*)(BIAS + mrow + 4);
    float bv[8] = {b0.x, b0.y, b0.z, b0.w, b1.x, b1.y, b1.z, b1.w};
    #pragma unroll
    for (int nt = 0; nt < 4; ++nt)
      #pragma unroll
      for (int r = 0; r < 8; ++r)
        acc[mt][nt][r] += bv[r];
  }

  const int tok0 = tblk * BT;
  #pragma unroll
  for (int nt = 0; nt < 4; ++nt) {
    const int trel = wn * 64 + nt * 16 + l16;
    const int tok  = tok0 + trel;
    const long long tv = TGT[tok];

    float mx = -3.0e38f;
    #pragma unroll
    for (int mt = 0; mt < 2; ++mt)
      #pragma unroll
      for (int r = 0; r < 8; ++r)
        mx = fmaxf(mx, acc[mt][nt][r]);

    float s = 0.f, tg = 0.f;
    bool found = false;
    #pragma unroll
    for (int mt = 0; mt < 2; ++mt) {
      const int mrow = vblk * BV + wm * 32 + mt * 16 + half * 8;
      #pragma unroll
      for (int r = 0; r < 8; ++r) {
        const float v = acc[mt][nt][r];
        s += exp2f((v - mx) * L2E);
        if ((long long)(mrow + r) == tv) { tg = v; found = true; }
      }
    }
    const int slot = wm * 2 + half;
    red_m[trel * 8 + slot] = mx;
    red_s[trel * 8 + slot] = s;
    if (found) tl[tok] = tg;
  }
  __syncthreads();

  if (tid < BT) {
    float M = -3.0e38f;
    #pragma unroll
    for (int j = 0; j < 8; ++j) M = fmaxf(M, red_m[tid * 8 + j]);
    float S = 0.f;
    #pragma unroll
    for (int j = 0; j < 8; ++j)
      S += red_s[tid * 8 + j] * exp2f((red_m[tid * 8 + j] - M) * L2E);
    const size_t o = (size_t)vblk * N_TOK + tok0 + tid;
    pm[o] = M;
    ps[o] = S;
  }
}

// ---------------------------------------------------------------------------
// f32 -> f16 pre-convert (vectorized elementwise)
// ---------------------------------------------------------------------------
__global__ __launch_bounds__(256) void cvt_f16(const float* __restrict__ src,
                                               _Float16* __restrict__ dst, int n4)
{
  const int i = blockIdx.x * 256 + threadIdx.x;
  if (i < n4) {
    float4 v = ((const float4*)src)[i];
    v4h h; h[0] = (_Float16)v.x; h[1] = (_Float16)v.y;
    h[2] = (_Float16)v.z; h[3] = (_Float16)v.w;
    ((v4h*)dst)[i] = h;
  }
}

// ---------------------------------------------------------------------------
// Fast main kernel: f16 operands, double-buffered ASYNC global->LDS staging.
// ---------------------------------------------------------------------------
__global__ __launch_bounds__(256) void fce_main_h(
    const _Float16* __restrict__ Wh, const _Float16* __restrict__ Xh,
    const long long* __restrict__ TGT, const float* __restrict__ BIAS,
    float* __restrict__ pm, float* __restrict__ ps, float* __restrict__ tl)
{
  __shared__ _Float16 sW[2][BV * LDK];
  __shared__ _Float16 sX[2][BT * LDK];
  __shared__ float red_m[BT * 8];
  __shared__ float red_s[BT * 8];

  const int tid  = threadIdx.x;
  const int wave = tid >> 5;
  const int lane = tid & 31;
  const int l16  = lane & 15;
  const int half = lane >> 4;

  const int vblk = blockIdx.x / TBLKS;
  const int tblk = blockIdx.x % TBLKS;
  const int wm = wave & 3;
  const int wn = wave >> 2;

  v8f acc[2][4];
  #pragma unroll
  for (int mt = 0; mt < 2; ++mt)
    #pragma unroll
    for (int nt = 0; nt < 4; ++nt)
      #pragma unroll
      for (int i = 0; i < 8; ++i) acc[mt][nt][i] = 0.0f;

  const size_t wbase = (size_t)vblk * BV * HID;
  const size_t xbase = (size_t)tblk * BT * HID;

  // Each thread moves 2 x 16B chunks per tile per matrix.
  // chunk c = tid*2 + j : row = c>>2 (of 128), sub = c&3 (16B sub-chunk of 64B row)
  const int c0 = tid * 2, c1 = c0 + 1;
  const int row0 = c0 >> 2, sub0 = c0 & 3;
  const int row1 = c1 >> 2, sub1 = c1 & 3;

  auto issue = [&](int k0, int b) {
    const uint32_t lw = lds_byte_off(&sW[b][0]);
    const uint32_t lx = lds_byte_off(&sX[b][0]);
    async_g2l_b128(lw + row0 * (LDK * 2) + sub0 * 16,
                   Wh + wbase + (size_t)row0 * HID + k0 + sub0 * 8);
    async_g2l_b128(lw + row1 * (LDK * 2) + sub1 * 16,
                   Wh + wbase + (size_t)row1 * HID + k0 + sub1 * 8);
    async_g2l_b128(lx + row0 * (LDK * 2) + sub0 * 16,
                   Xh + xbase + (size_t)row0 * HID + k0 + sub0 * 8);
    async_g2l_b128(lx + row1 * (LDK * 2) + sub1 * 16,
                   Xh + xbase + (size_t)row1 * HID + k0 + sub1 * 8);
  };

  issue(0, 0);
  for (int i = 0; i < NKIT; ++i) {
    const int b = i & 1;
    wait_async0();          // my chunks of tile i are in LDS
    __syncthreads();        // everyone's chunks visible; prev reads of buf 1-b done
    if (i + 1 < NKIT) issue((i + 1) * BK, 1 - b);   // prefetch next tile
    do_tile(&sW[b][0], &sX[b][0], wm, wn, l16, half, acc);
  }

  epilogue(acc, vblk, tblk, wm, wn, l16, half, tid, TGT, BIAS, pm, ps, tl, red_m, red_s);
}

// ---------------------------------------------------------------------------
// Fallback main kernel: f32 operands, in-kernel conversion (batched loads).
// ---------------------------------------------------------------------------
__global__ __launch_bounds__(256) void fce_main_f(
    const float* __restrict__ W, const float* __restrict__ X,
    const long long* __restrict__ TGT, const float* __restrict__ BIAS,
    float* __restrict__ pm, float* __restrict__ ps, float* __restrict__ tl)
{
  __shared__ _Float16 sW[BV * LDK];
  __shared__ _Float16 sX[BT * LDK];
  __shared__ float red_m[BT * 8];
  __shared__ float red_s[BT * 8];

  const int tid  = threadIdx.x;
  const int wave = tid >> 5;
  const int lane = tid & 31;
  const int l16  = lane & 15;
  const int half = lane >> 4;

  const int vblk = blockIdx.x / TBLKS;
  const int tblk = blockIdx.x % TBLKS;
  const int wm = wave & 3;
  const int wn = wave >> 2;

  v8f acc[2][4];
  #pragma unroll
  for (int mt = 0; mt < 2; ++mt)
    #pragma unroll
    for (int nt = 0; nt < 4; ++nt)
      #pragma unroll
      for (int i = 0; i < 8; ++i) acc[mt][nt][i] = 0.0f;

  const size_t wbase = (size_t)vblk * BV * HID;
  const size_t xbase = (size_t)tblk * BT * HID;

  for (int k0 = 0; k0 < HID; k0 += BK) {
    __syncthreads();
    float4 wv[4], xv[4];
    #pragma unroll
    for (int i = 0; i < 4; ++i) {                  // batch all loads first
      const int e = (i * 256 + tid) * 4;
      const int r = e >> 5;
      const int c = e & 31;
      wv[i] = *(const float4*)(W + wbase + (size_t)r * HID + k0 + c);
      xv[i] = *(const float4*)(X + xbase + (size_t)r * HID + k0 + c);
    }
    #pragma unroll
    for (int i = 0; i < 4; ++i) {                  // then convert + store
      const int e = (i * 256 + tid) * 4;
      const int r = e >> 5;
      const int c = e & 31;
      _Float16* dw = &sW[r * LDK + c];
      dw[0] = (_Float16)wv[i].x; dw[1] = (_Float16)wv[i].y;
      dw[2] = (_Float16)wv[i].z; dw[3] = (_Float16)wv[i].w;
      _Float16* dx = &sX[r * LDK + c];
      dx[0] = (_Float16)xv[i].x; dx[1] = (_Float16)xv[i].y;
      dx[2] = (_Float16)xv[i].z; dx[3] = (_Float16)xv[i].w;
    }
    __syncthreads();
    if (k0 + BK < HID)
      __builtin_prefetch(W + wbase + (size_t)(tid >> 1) * HID + k0 + BK + (tid & 1) * 16, 0, 0);
    do_tile(sW, sX, wm, wn, l16, half, acc);
  }

  epilogue(acc, vblk, tblk, wm, wn, l16, half, tid, TGT, BIAS, pm, ps, tl, red_m, red_s);
}

// ---------------------------------------------------------------------------
// Pass 2: per-token reduction over vocab-block partials -> mean nll
// ---------------------------------------------------------------------------
__global__ __launch_bounds__(256) void fce_reduce(
    const float* __restrict__ pm, const float* __restrict__ ps,
    const float* __restrict__ tl, const long long* __restrict__ TGT,
    float* __restrict__ accum)
{
  __shared__ float rs[256];
  __shared__ float rc[256];
  const int t = blockIdx.x * 256 + threadIdx.x;

  float M = -3.0e38f;
  for (int j = 0; j < VBLKS; ++j)
    M = fmaxf(M, pm[(size_t)j * N_TOK + t]);
  float S = 0.f;
  for (int j = 0; j < VBLKS; ++j)
    S += ps[(size_t)j * N_TOK + t] * exp2f((pm[(size_t)j * N_TOK + t] - M) * L2E);

  const float lse = M + LN2 * log2f(S);
  const long long tv = TGT[t];
  const bool valid = (tv != -100LL);
  rs[threadIdx.x] = valid ? (lse - tl[t]) : 0.f;
  rc[threadIdx.x] = valid ? 1.f : 0.f;
  __syncthreads();
  for (int off = 128; off > 0; off >>= 1) {
    if (threadIdx.x < off) {
      rs[threadIdx.x] += rs[threadIdx.x + off];
      rc[threadIdx.x] += rc[threadIdx.x + off];
    }
    __syncthreads();
  }
  if (threadIdx.x == 0) {
    atomicAdd(&accum[0], rs[0]);
    atomicAdd(&accum[1], rc[0]);
  }
}

__global__ void fce_final(const float* __restrict__ accum, float* __restrict__ out)
{
  out[0] = accum[0] / fmaxf(accum[1], 1.0f);
}

// ---------------------------------------------------------------------------
extern "C" void kernel_launch(void* const* d_in, const int* in_sizes, int n_in,
                              void* d_out, int out_size, void* d_ws, size_t ws_size,
                              hipStream_t stream)
{
  (void)in_sizes; (void)n_in; (void)out_size;
  const float*     W   = (const float*)d_in[0];      // [VOC, HID] f32
  const float*     X   = (const float*)d_in[1];      // [N_TOK, HID] f32
  const long long* TGT = (const long long*)d_in[2];  // [N_TOK] i64
  const float*     B   = (const float*)d_in[3];      // [VOC] f32

  float* pm    = (float*)d_ws;                       // [VBLKS][N_TOK]
  float* ps    = pm + (size_t)VBLKS * N_TOK;         // [VBLKS][N_TOK]
  float* tl    = ps + (size_t)VBLKS * N_TOK;         // [N_TOK]
  float* accum = tl + N_TOK;                         // [4] (sum, count, pad)

  const size_t part_bytes = ((size_t)2 * VBLKS * N_TOK + N_TOK + 4) * sizeof(float);
  const size_t half_bytes = ((size_t)VOC * HID + (size_t)N_TOK * HID) * sizeof(_Float16);

  hipMemsetAsync(accum, 0, 2 * sizeof(float), stream);

  if (ws_size >= part_bytes + half_bytes) {
    // Pre-convert to f16 in workspace, then async-staged WMMA GEMM.
    _Float16* Wh = (_Float16*)((char*)d_ws + part_bytes);
    _Float16* Xh = Wh + (size_t)VOC * HID;
    const int wn4 = (VOC * HID) / 4;
    const int xn4 = (N_TOK * HID) / 4;
    cvt_f16<<<dim3((wn4 + 255) / 256), dim3(256), 0, stream>>>(W, Wh, wn4);
    cvt_f16<<<dim3((xn4 + 255) / 256), dim3(256), 0, stream>>>(X, Xh, xn4);
    fce_main_h<<<dim3(VBLKS * TBLKS), dim3(256), 0, stream>>>(Wh, Xh, TGT, B, pm, ps, tl);
  } else {
    fce_main_f<<<dim3(VBLKS * TBLKS), dim3(256), 0, stream>>>(W, X, TGT, B, pm, ps, tl);
  }

  fce_reduce<<<dim3(N_TOK / 256), dim3(256), 0, stream>>>(pm, ps, tl, TGT, accum);
  fce_final<<<1, 1, 0, stream>>>(accum, (float*)d_out);
}
